// Block_76862734729555
// MI455X (gfx1250) — compile-verified, matrix operand order checked by reading, not run on hardware
//
#include <hip/hip_runtime.h>
#include <hip/hip_bf16.h>

// ---------------------------------------------------------------------------
// Transformer block for MI455X (gfx1250, wave32, WMMA bf16 16x16x32).
//   x = x + attn(rms(x));  h = relu(bank_fc(rms(x)))^2;  x = x + bank_pj(h)
// All GEMM math goes through v_wmma_f32_16x16x32_bf16 (fp32 accum).
// GEMM W-tiles are staged into double-buffered LDS via CDNA5 async paths:
//   MODE 0: global_load_async_to_lds_b128  (ASYNCcnt,  per-lane)
//   MODE 1: tensor_load_to_lds (TDM D# descriptor, TENSORcnt, one per tile)
// Workspace requirement ~= 280 MB (phase-overlapped union layout below).
// ---------------------------------------------------------------------------

typedef unsigned short u16;
typedef __attribute__((ext_vector_type(16))) __bf16 v16bf;
typedef __attribute__((ext_vector_type(8)))  float  v8f;
typedef __attribute__((ext_vector_type(4)))  unsigned u32x4;
typedef __attribute__((ext_vector_type(4)))  unsigned tdm_g0_t;
typedef __attribute__((ext_vector_type(8)))  int      tdm_g1_t;
typedef __attribute__((ext_vector_type(4)))  int      tdm_g2_t;

constexpr int Bc   = 2;
constexpr int TTc  = 2048;
constexpr int DIMc = 1024;
constexpr int Hc   = 16;
constexpr int HKVc = 4;
constexpr int HDc  = 64;
constexpr int HIDc = 4096;
constexpr int KVc  = HKVc * HDc;       // 256
constexpr int MMc  = Bc * TTc;         // 4096 rows

// ---- helpers --------------------------------------------------------------

__device__ __forceinline__ u16 f2bf(float f) {
  unsigned u = __float_as_uint(f);
  unsigned r = u + 0x7fffu + ((u >> 16) & 1u);   // round-to-nearest-even
  return (u16)(r >> 16);
}

// Load one 16x32 bf16 A/B fragment (wave32 layout, ISA 7.12.2):
//   lane<16 : row = lane,     K-runs [0..7] and [16..23]
//   lane>=16: row = lane-16,  K-runs [8..15] and [24..31]
__device__ __forceinline__ v16bf frag_ab(const u16* base, int ld) {
  int lane = threadIdx.x & 31;
  const u16* p = base + (size_t)(lane & 15) * ld + ((lane < 16) ? 0 : 8);
  union FU { u32x4 q[2]; v16bf v; } f;
  f.q[0] = *(const u32x4*)(p);
  f.q[1] = *(const u32x4*)(p + 16);
  return f.v;
}

__device__ __forceinline__ v8f wmma_bf16(v16bf a, v16bf b, v8f c) {
  return __builtin_amdgcn_wmma_f32_16x16x32_bf16(
      false, a, false, b, (short)0, c, false, false);
}

// per-lane async memory->LDS copy, 16 bytes/lane (GLOBAL_LOAD_ASYNC_TO_LDS_B128)
__device__ __forceinline__ void async_load_b128(unsigned lds_off, const u16* gptr) {
  asm volatile("global_load_async_to_lds_b128 %0, %1, off"
               :: "v"(lds_off), "v"(gptr) : "memory");
}
__device__ __forceinline__ void wait_async0() {
  asm volatile("s_wait_asynccnt 0x0" ::: "memory");
}

// TDM: one descriptor moves a (tile_h x tile_w) bf16 tile (row stride =
// stride_elems) from global memory into LDS at lds_off.  D# packing per
// CDNA5 ISA 8.3/8.4: group0 = {count=1, lds_addr, global_addr, type=2},
// group1 = {data_size=2B, tensor/tile dims, dim0 stride}.  2D -> groups 2/3
// zero (VADDR2/3 semantics: unused dims).
__device__ __forceinline__ void tdm_load_2d(unsigned lds_off, const void* gaddr,
                                            unsigned tile_w, unsigned tile_h,
                                            long stride_elems) {
  unsigned long long ga = (unsigned long long)gaddr;
  tdm_g0_t g0;
  g0[0] = 1u;                                           // count=1 (valid D#)
  g0[1] = lds_off;                                      // lds_addr (bytes)
  g0[2] = (unsigned)(ga & 0xffffffffu);                 // global_addr[31:0]
  g0[3] = (unsigned)((ga >> 32) & 0x01ffffffu) | (2u << 30);  // [56:32] | type=2
  tdm_g1_t g1;
  g1[0] = 0x00010000;                                   // data_size=1 (2 bytes)
  g1[1] = (int)(tile_w << 16);                          // tensor_dim0[15:0]
  g1[2] = (int)((tile_w >> 16) | (tile_h << 16));       // td0 hi | tensor_dim1 lo
  g1[3] = (int)((tile_h >> 16) | (tile_w << 16));       // td1 hi | tile_dim0
  g1[4] = (int)(tile_h & 0xffffu);                      // tile_dim1 (tile_dim2=0)
  g1[5] = (int)(stride_elems & 0xffffffff);             // tensor_dim0_stride lo
  g1[6] = (int)((stride_elems >> 32) & 0xffff);         // stride hi | d1_stride=0
  g1[7] = 0;
  tdm_g2_t z4 = {0, 0, 0, 0};
#if defined(__clang_major__) && (__clang_major__ >= 23)
  tdm_g1_t z8 = {0, 0, 0, 0, 0, 0, 0, 0};
  __builtin_amdgcn_tensor_load_to_lds(g0, g1, z4, z4, z8, 0);
#else
  __builtin_amdgcn_tensor_load_to_lds(g0, g1, z4, z4, 0);
#endif
}

// ---- tiny kernels ---------------------------------------------------------

__global__ void k_topk(const float* fl, const float* pl, int* idx, float* coef) {
  if (threadIdx.x || blockIdx.x) return;
  const float* L[2] = {fl, pl};
  for (int s = 0; s < 2; ++s) {
    int i0 = 0;
    for (int i = 1; i < 8; ++i) if (L[s][i] > L[s][i0]) i0 = i;
    int i1 = -1;
    for (int i = 0; i < 8; ++i) {
      if (i == i0) continue;
      if (i1 < 0 || L[s][i] > L[s][i1]) i1 = i;
    }
    float v0 = L[s][i0], v1 = L[s][i1];
    float m = fmaxf(v0, v1);
    float e0 = __expf(v0 - m), e1 = __expf(v1 - m), d = e0 + e1;
    idx[s * 2 + 0] = i0; idx[s * 2 + 1] = i1;
    coef[s * 2 + 0] = e0 / d; coef[s * 2 + 1] = e1 / d;
  }
}

__global__ void k_zero(float* p, long n) {
  long i = (long)blockIdx.x * blockDim.x + threadIdx.x;
  long st = (long)gridDim.x * blockDim.x;
  for (; i < n; i += st) p[i] = 0.0f;
}

__global__ void k_cvt(const float* w, u16* o, long n) {
  long i = (long)blockIdx.x * blockDim.x + threadIdx.x;
  long st = (long)gridDim.x * blockDim.x;
  for (; i < n; i += st) o[i] = f2bf(w[i]);
}

__global__ void k_cvt_bank(const float* w, u16* o, const int* idx, int which, long per) {
  const float* b = w + (long)idx[which] * per;
  long i = (long)blockIdx.x * blockDim.x + threadIdx.x;
  long st = (long)gridDim.x * blockDim.x;
  for (; i < per; i += st) o[i] = f2bf(b[i]);
}

// blockdiag weights: (g, c, d) fp32 -> (g, d, c) bf16 ((N,K)-major blocks)
__global__ void k_cvt_blkT(const float* s, u16* o, const int* idx, int which, int nb) {
  long n = (long)nb * 4096;
  const float* b = s + (long)idx[which] * n;
  long i = (long)blockIdx.x * blockDim.x + threadIdx.x;
  long st = (long)gridDim.x * blockDim.x;
  for (; i < n; i += st) {
    long g = i >> 12;
    int  r = (int)(i & 4095);
    int  d = r >> 6, c = r & 63;
    o[i] = f2bf(b[(g << 12) + (c << 6) + d]);
  }
}

__global__ void k_rms(const float* x, u16* o, int dim) {
  int row = blockIdx.x, tid = threadIdx.x;
  const float* xr = x + (size_t)row * dim;
  float s = 0.0f;
  for (int j = tid; j < dim; j += 256) { float v = xr[j]; s += v * v; }
#pragma unroll
  for (int m = 16; m; m >>= 1) s += __shfl_xor(s, m, 32);
  __shared__ float red[8];
  if ((tid & 31) == 0) red[tid >> 5] = s;
  __syncthreads();
  float tot = 0.0f;
#pragma unroll
  for (int w = 0; w < 8; ++w) tot += red[w];
  float rn = rsqrtf(tot / dim + 1e-6f);
  for (int j = tid; j < dim; j += 256) o[(size_t)row * dim + j] = f2bf(xr[j] * rn);
}

// out[m, c*n_in+g] = in[m, ((g-shift) mod n_in)*64 + c]
__global__ void k_roll_swap(const float* t1, u16* t2, const int* idx, int which,
                            int n_in, long n) {
  int F = n_in * 64;
  int shift = idx[which] % n_in;
  long i = (long)blockIdx.x * blockDim.x + threadIdx.x;
  long st = (long)gridDim.x * blockDim.x;
  for (; i < n; i += st) {
    long m = i / F;
    int  j = (int)(i - m * F);
    int  c = j / n_in, g = j - c * n_in;
    int  gs = g - shift; if (gs < 0) gs += n_in;
    t2[i] = f2bf(t1[m * F + gs * 64 + c]);
  }
}

__global__ void k_relu2(const float* a, u16* o, long n) {
  long i = (long)blockIdx.x * blockDim.x + threadIdx.x;
  long st = (long)gridDim.x * blockDim.x;
  for (; i < n; i += st) { float v = fmaxf(a[i], 0.0f); o[i] = f2bf(v * v); }
}

__global__ void k_addout(const float* x1, const float* a, float* out, long n) {
  long i = (long)blockIdx.x * blockDim.x + threadIdx.x;
  long st = (long)gridDim.x * blockDim.x;
  for (; i < n; i += st) out[i] = x1[i] + a[i];
}

__global__ void k_qpost(const float* qf, const float* qg, u16* qr) {
  int wave = threadIdx.x >> 5, lane = threadIdx.x & 31;
  int gid = blockIdx.x * 4 + wave;            // row*H + h
  int row = gid >> 4, h = gid & 15;
  int b = row / TTc, t = row - b * TTc;
  const float* p = qf + (size_t)row * DIMc + h * HDc;
  float a = p[lane], c2 = p[lane + 32];
  float s = a * a + c2 * c2;
#pragma unroll
  for (int m = 16; m; m >>= 1) s += __shfl_xor(s, m, 32);
  float rn = rsqrtf(s / 64.0f + 1e-6f) * qg[h];
  a *= rn; c2 *= rn;
  float fr = (float)t * __powf(10000.0f, -(float)lane / 32.0f);
  float cs = __cosf(fr), sn = __sinf(fr);
  u16* out = qr + (((size_t)(b * Hc + h)) * TTc + t) * HDc;
  out[lane]      = f2bf(a * cs + c2 * sn);
  out[lane + 32] = f2bf(-a * sn + c2 * cs);
}

__global__ void k_kvpost(const float* kf, const float* vf, const float* kg,
                         u16* kr, u16* vt) {
  int wave = threadIdx.x >> 5, lane = threadIdx.x & 31;
  int gid = blockIdx.x * 4 + wave;            // row*HKV + h
  int row = gid >> 2, h = gid & 3;
  int b = row / TTc, t = row - b * TTc;
  const float* p = kf + (size_t)row * KVc + h * HDc;
  float a = p[lane], c2 = p[lane + 32];
  float s = a * a + c2 * c2;
#pragma unroll
  for (int m = 16; m; m >>= 1) s += __shfl_xor(s, m, 32);
  float rn = rsqrtf(s / 64.0f + 1e-6f) * kg[h];
  a *= rn; c2 *= rn;
  float fr = (float)t * __powf(10000.0f, -(float)lane / 32.0f);
  float cs = __cosf(fr), sn = __sinf(fr);
  u16* outk = kr + (((size_t)(b * HKVc + h)) * TTc + t) * HDc;
  outk[lane]      = f2bf(a * cs + c2 * sn);
  outk[lane + 32] = f2bf(-a * sn + c2 * cs);
  const float* pv = vf + (size_t)row * KVc + h * HDc;
  u16* outv = vt + ((size_t)(b * HKVc + h)) * HDc * TTc;
  outv[(size_t)lane * TTc + t]        = f2bf(pv[lane]);
  outv[(size_t)(lane + 32) * TTc + t] = f2bf(pv[lane + 32]);
}

// ---- WMMA GEMM:  C[M,N](f32 and/or bf16) = A[M,K]bf16 * W[N,K]bf16^T ------
// 4 waves/WG, 64x64 tile, double-buffered LDS W-tile with async staging.
// MODE 0: per-lane GLOBAL_LOAD_ASYNC_TO_LDS_B128;  MODE 1: TDM descriptor.
template <int MODE>
__global__ void k_gemm(const u16* __restrict__ A, const u16* __restrict__ W,
                       float* C, const float* res, u16* Cb,
                       int Mm, int Nn, int K) {
  __shared__ u16 sw[2][64 * 32];
  int m0 = blockIdx.x * 64, n0 = blockIdx.y * 64;
  int wave = threadIdx.x >> 5, lane = threadIdx.x & 31;
  int mw = m0 + wave * 16;
  v8f acc[4] = {};

  auto stage = [&](int buf, int kb) {
    if constexpr (MODE == 0) {
      int t = threadIdx.x;
#pragma unroll
      for (int it = 0; it < 2; ++it) {
        int id = t + it * 128;                 // 256 * 16B = 64x32 bf16
        int nn = id >> 2, kk2 = (id & 3) * 8;
        async_load_b128((unsigned)(size_t)&sw[buf][nn * 32 + kk2],
                        &W[(size_t)(n0 + nn) * K + kb + kk2]);
      }
    } else {
      if (threadIdx.x < 32)                    // wave 0 issues one TDM D#
        tdm_load_2d((unsigned)(size_t)&sw[buf][0], &W[(size_t)n0 * K + kb],
                    32u, 64u, (long)K);
    }
  };

  stage(0, 0);
  for (int kb = 0; kb < K; kb += 32) {
    int cur = (kb >> 5) & 1;
    if constexpr (MODE == 0) {
      wait_async0();
    } else {
      if (threadIdx.x < 32) __builtin_amdgcn_s_wait_tensorcnt(0);
    }
    __syncthreads();                           // tile `cur` ready for all waves
    if (kb + 32 < K) stage(cur ^ 1, kb + 32);  // overlap next tile with WMMAs
    v16bf af = frag_ab(&A[(size_t)mw * K + kb], K);
#pragma unroll
    for (int nt = 0; nt < 4; ++nt)
      acc[nt] = wmma_bf16(af, frag_ab(&sw[cur][(nt * 16) * 32], 32), acc[nt]);
  }
  int half = lane >> 4, l15 = lane & 15;
#pragma unroll
  for (int nt = 0; nt < 4; ++nt)
#pragma unroll
    for (int r = 0; r < 8; ++r) {
      int m = mw + r + 8 * half;
      size_t off = (size_t)m * Nn + n0 + nt * 16 + l15;
      float v = acc[nt][r];
      if (res) v += res[off];
      if (C)  C[off] = v;
      if (Cb) Cb[off] = f2bf(v);
    }
}

// ---- WMMA block-diagonal:  out[m, g*64+d] (+)= sum_c A[m,g*64+c]*Wt[g,d,c]
template <bool ACC>
__global__ void k_bdiag(const u16* __restrict__ A, const u16* __restrict__ Wt,
                        float* Out, const float* coefp, int cidx, int nb) {
  int wave = threadIdx.x >> 5, lane = threadIdx.x & 31;
  int m0 = (blockIdx.x * 4 + wave) * 16;
  int g = blockIdx.y;
  int K = nb * 64;
  const u16* Ab = A + (size_t)m0 * K + g * 64;
  const u16* Wb = Wt + (size_t)g * 4096;
  v8f acc[4] = {};
#pragma unroll
  for (int kb = 0; kb < 64; kb += 32) {
    v16bf af = frag_ab(Ab + kb, K);
#pragma unroll
    for (int nt = 0; nt < 4; ++nt)
      acc[nt] = wmma_bf16(af, frag_ab(Wb + (size_t)(nt * 16) * 64 + kb, 64), acc[nt]);
  }
  float coef = ACC ? coefp[cidx] : 1.0f;
  int half = lane >> 4, l15 = lane & 15;
#pragma unroll
  for (int nt = 0; nt < 4; ++nt)
#pragma unroll
    for (int r = 0; r < 8; ++r) {
      size_t off = (size_t)(m0 + r + 8 * half) * K + g * 64 + nt * 16 + l15;
      if (ACC) Out[off] += coef * acc[nt][r];
      else     Out[off] = acc[nt][r];
    }
}

// ---- flash attention (causal, GQA), fully in WMMA -------------------------
__global__ void k_flash(const u16* __restrict__ qr, const u16* __restrict__ kr,
                        const u16* __restrict__ vt, u16* __restrict__ yb) {
  __shared__ u16 sp[4][16 * 32];               // per-wave P tile (bf16)
  int wave = threadIdx.x >> 5, lane = threadIdx.x & 31;
  int gid = blockIdx.x * 4 + wave;             // over B*H*(T/16)
  int i = gid & 127, bh = gid >> 7;
  int h = bh & 15, b = bh >> 4;
  int hk = h >> 2;                             // H/HKV = 4
  const u16* qbase = qr + ((size_t)(b * Hc + h) * TTc + i * 16) * HDc;
  const u16* kbase = kr + ((size_t)(b * HKVc + hk) * TTc) * HDc;
  const u16* vbase = vt + ((size_t)(b * HKVc + hk)) * HDc * TTc;
  u16* pw = &sp[wave][0];
  v16bf qa0 = frag_ab(qbase + 0, HDc);
  v16bf qa1 = frag_ab(qbase + 32, HDc);
  v8f o[4] = {};
  float mrow[8], lrow[8];
#pragma unroll
  for (int r = 0; r < 8; ++r) { mrow[r] = -1e30f; lrow[r] = 0.0f; }
  int half = lane >> 4, l15 = lane & 15;
  int rowg0 = i * 16;
  int nch = (i * 16 + 16 + 31) >> 5;           // 32-key chunks (causal bound)
  for (int j = 0; j < nch; ++j) {
    int k0 = j * 32;
    float sm[2][8];
    float cmax[8];
#pragma unroll
    for (int r = 0; r < 8; ++r) cmax[r] = -1e30f;
#pragma unroll
    for (int jj = 0; jj < 2; ++jj) {
      const u16* kb2 = kbase + (size_t)(k0 + jj * 16) * HDc;
      v8f s = {};
      s = wmma_bf16(qa0, frag_ab(kb2 + 0, HDc), s);
      s = wmma_bf16(qa1, frag_ab(kb2 + 32, HDc), s);
      int colg = k0 + jj * 16 + l15;
#pragma unroll
      for (int r = 0; r < 8; ++r) {
        float v = s[r] * 0.125f;               // 1/sqrt(HD)
        int rowg = rowg0 + r + 8 * half;
        if (colg > rowg) v = -1e30f;           // causal mask
        sm[jj][r] = v;
        cmax[r] = fmaxf(cmax[r], v);
      }
    }
#pragma unroll
    for (int r = 0; r < 8; ++r) {
#pragma unroll
      for (int m = 1; m < 16; m <<= 1)
        cmax[r] = fmaxf(cmax[r], __shfl_xor(cmax[r], m, 32));
    }
    float al[8];
#pragma unroll
    for (int r = 0; r < 8; ++r) {
      float nm = fmaxf(mrow[r], cmax[r]);
      al[r] = __expf(mrow[r] - nm);
      mrow[r] = nm;
    }
    float rs[8];
#pragma unroll
    for (int r = 0; r < 8; ++r) rs[r] = 0.0f;
#pragma unroll
    for (int jj = 0; jj < 2; ++jj)
#pragma unroll
      for (int r = 0; r < 8; ++r) {
        float p = __expf(sm[jj][r] - mrow[r]);
        rs[r] += p;
        pw[(r + 8 * half) * 32 + jj * 16 + l15] = f2bf(p);  // C-layout -> LDS
      }
#pragma unroll
    for (int r = 0; r < 8; ++r) {
#pragma unroll
      for (int m = 1; m < 16; m <<= 1) rs[r] += __shfl_xor(rs[r], m, 32);
      lrow[r] = lrow[r] * al[r] + rs[r];
    }
#pragma unroll
    for (int nt = 0; nt < 4; ++nt)
#pragma unroll
      for (int r = 0; r < 8; ++r) o[nt][r] *= al[r];
    asm volatile("s_wait_dscnt 0" ::: "memory");   // P visible for re-load
    v16bf pa = frag_ab(pw, 32);                    // LDS -> A-fragment
#pragma unroll
    for (int nt = 0; nt < 4; ++nt)
      o[nt] = wmma_bf16(pa, frag_ab(vbase + (size_t)(nt * 16) * TTc + k0, TTc),
                        o[nt]);
  }
#pragma unroll
  for (int nt = 0; nt < 4; ++nt)
#pragma unroll
    for (int r = 0; r < 8; ++r) {
      int rowg = rowg0 + r + 8 * half;
      yb[((size_t)(b * TTc) + rowg) * DIMc + h * HDc + nt * 16 + l15] =
          f2bf(o[nt][r] / lrow[r]);
    }
}

// ---------------------------------------------------------------------------

extern "C" void kernel_launch(void* const* d_in, const int* in_sizes, int n_in,
                              void* d_out, int out_size, void* d_ws, size_t ws_size,
                              hipStream_t stream) {
  (void)in_sizes; (void)n_in; (void)out_size; (void)ws_size;
  const float* x         = (const float*)d_in[0];
  const float* wq        = (const float*)d_in[1];
  const float* wk        = (const float*)d_in[2];
  const float* wv        = (const float*)d_in[3];
  const float* wo        = (const float*)d_in[4];
  const float* qg        = (const float*)d_in[5];
  const float* kg        = (const float*)d_in[6];
  const float* fc_s1     = (const float*)d_in[7];
  const float* fc_s2     = (const float*)d_in[8];
  const float* fc_mix    = (const float*)d_in[9];
  const float* fc_logits = (const float*)d_in[10];
  const float* pj_s1     = (const float*)d_in[11];
  const float* pj_s2     = (const float*)d_in[12];
  const float* pj_mix    = (const float*)d_in[13];
  const float* pj_logits = (const float*)d_in[14];

  char* ws = (char*)d_ws;
  size_t cur = 0;
  auto alloc = [&](size_t bytes) -> char* {
    cur = (cur + 255) & ~(size_t)255;
    char* p = ws + cur;
    cur += bytes;
    return p;
  };

  int*   idx  = (int*)alloc(256);
  float* coef = (float*)((char*)idx + 128);
  float* x1   = (float*)alloc((size_t)MMc * DIMc * 4);
  size_t unionBase = cur;

  // ---- phase A (attention) layout ----
  cur = unionBase;
  u16* wqb = (u16*)alloc((size_t)DIMc * DIMc * 2);
  u16* wkb = (u16*)alloc((size_t)KVc * DIMc * 2);
  u16* wvb = (u16*)alloc((size_t)KVc * DIMc * 2);
  u16* wob = (u16*)alloc((size_t)DIMc * DIMc * 2);
  u16* xnb = (u16*)alloc((size_t)MMc * DIMc * 2);
  float* qf = (float*)alloc((size_t)MMc * DIMc * 4);
  float* kf = (float*)alloc((size_t)MMc * KVc * 4);
  float* vf = (float*)alloc((size_t)MMc * KVc * 4);
  u16* qrp = (u16*)alloc((size_t)MMc * DIMc * 2);
  u16* krp = (u16*)alloc((size_t)MMc * KVc * 2);
  u16* vtp = (u16*)alloc((size_t)MMc * KVc * 2);
  u16* yb  = (u16*)alloc((size_t)MMc * DIMc * 2);

  // ---- phase B (banked MLP) layout (overlaps phase A) ----
  cur = unionBase;
  u16* hb    = (u16*)alloc((size_t)MMc * DIMc * 2);
  u16* s1t   = (u16*)alloc((size_t)64 * 4096 * 2);
  u16* s2t   = (u16*)alloc((size_t)64 * 4096 * 2);
  u16* mixb  = (u16*)alloc((size_t)HIDc * DIMc * 2);
  float* t1  = (float*)alloc((size_t)MMc * HIDc * 4);
  u16* t2b   = (u16*)alloc((size_t)MMc * HIDc * 2);
  u16* t3b   = (u16*)alloc((size_t)MMc * HIDc * 2);
  float* accfc = (float*)alloc((size_t)MMc * HIDc * 4);
  u16* hmid  = (u16*)alloc((size_t)MMc * HIDc * 2);
  float* accpj = (float*)alloc((size_t)MMc * DIMc * 4);

  auto GB = [](long n) { return dim3((unsigned)((n + 255) / 256)); };

  // ---- phase A ----
  k_topk<<<1, 32, 0, stream>>>(fc_logits, pj_logits, idx, coef);
  k_rms<<<MMc, 256, 0, stream>>>(x, xnb, DIMc);
  k_cvt<<<GB((long)DIMc * DIMc), 256, 0, stream>>>(wq, wqb, (long)DIMc * DIMc);
  k_cvt<<<GB((long)KVc * DIMc), 256, 0, stream>>>(wk, wkb, (long)KVc * DIMc);
  k_cvt<<<GB((long)KVc * DIMc), 256, 0, stream>>>(wv, wvb, (long)KVc * DIMc);
  k_cvt<<<GB((long)DIMc * DIMc), 256, 0, stream>>>(wo, wob, (long)DIMc * DIMc);

  k_gemm<0><<<dim3(MMc / 64, DIMc / 64), 128, 0, stream>>>(xnb, wqb, qf, nullptr, nullptr, MMc, DIMc, DIMc);
  k_gemm<0><<<dim3(MMc / 64, KVc / 64), 128, 0, stream>>>(xnb, wkb, kf, nullptr, nullptr, MMc, KVc, DIMc);
  k_gemm<0><<<dim3(MMc / 64, KVc / 64), 128, 0, stream>>>(xnb, wvb, vf, nullptr, nullptr, MMc, KVc, DIMc);

  k_qpost<<<MMc * Hc / 4, 128, 0, stream>>>(qf, qg, qrp);
  k_kvpost<<<MMc * HKVc / 4, 128, 0, stream>>>(kf, vf, kg, krp, vtp);

  k_flash<<<Bc * Hc * (TTc / 16) / 4, 128, 0, stream>>>(qrp, krp, vtp, yb);

  k_gemm<0><<<dim3(MMc / 64, DIMc / 64), 128, 0, stream>>>(yb, wob, x1, x, nullptr, MMc, DIMc, DIMc);

  // ---- phase B: fc bank_linear -> relu^2 -> pj bank_linear ----
  k_rms<<<MMc, 256, 0, stream>>>(x1, hb, DIMc);
  k_zero<<<GB((long)MMc * HIDc), 256, 0, stream>>>(accfc, (long)MMc * HIDc);
  for (int kk = 0; kk < 2; ++kk) {
    k_cvt_blkT<<<GB((long)16 * 4096), 256, 0, stream>>>(fc_s1, s1t, idx, kk, 16);
    k_cvt_blkT<<<GB((long)64 * 4096), 256, 0, stream>>>(fc_s2, s2t, idx, kk, 64);
    k_cvt_bank<<<GB((long)HIDc * DIMc), 256, 0, stream>>>(fc_mix, mixb, idx, kk, (long)HIDc * DIMc);
    k_bdiag<false><<<dim3(MMc / 64, 16), 128, 0, stream>>>(hb, s1t, t1, nullptr, 0, 16);
    k_roll_swap<<<GB((long)MMc * DIMc), 256, 0, stream>>>(t1, t2b, idx, kk, 16, (long)MMc * DIMc);
    k_gemm<1><<<dim3(MMc / 64, HIDc / 64), 128, 0, stream>>>(t2b, mixb, nullptr, nullptr, t3b, MMc, HIDc, DIMc);
    k_bdiag<true><<<dim3(MMc / 64, 64), 128, 0, stream>>>(t3b, s2t, accfc, coef, kk, 64);
  }
  k_relu2<<<GB((long)MMc * HIDc), 256, 0, stream>>>(accfc, hmid, (long)MMc * HIDc);

  k_zero<<<GB((long)MMc * DIMc), 256, 0, stream>>>(accpj, (long)MMc * DIMc);
  for (int kk = 0; kk < 2; ++kk) {
    k_cvt_blkT<<<GB((long)64 * 4096), 256, 0, stream>>>(pj_s1, s1t, idx, 2 + kk, 64);
    k_cvt_blkT<<<GB((long)16 * 4096), 256, 0, stream>>>(pj_s2, s2t, idx, 2 + kk, 16);
    k_cvt_bank<<<GB((long)DIMc * HIDc), 256, 0, stream>>>(pj_mix, mixb, idx, 2 + kk, (long)DIMc * HIDc);
    k_bdiag<false><<<dim3(MMc / 64, 64), 128, 0, stream>>>(hmid, s1t, t1, nullptr, 0, 64);
    k_roll_swap<<<GB((long)MMc * HIDc), 256, 0, stream>>>(t1, t2b, idx, 2 + kk, 64, (long)MMc * HIDc);
    k_gemm<1><<<dim3(MMc / 64, DIMc / 64), 128, 0, stream>>>(t2b, mixb, nullptr, nullptr, t3b, MMc, DIMc, HIDc);
    k_bdiag<true><<<dim3(MMc / 64, 16), 128, 0, stream>>>(t3b, s2t, accpj, coef, 2 + kk, 16);
  }

  k_addout<<<GB((long)MMc * DIMc), 256, 0, stream>>>(x1, accpj, (float*)d_out, (long)MMc * DIMc);
}
